// TransImg_23261542875590
// MI455X (gfx1250) — matrix operand order repaired and checked by reading, not compiled, not run on hardware
//
#include <hip/hip_runtime.h>
#include <hip/hip_bf16.h>

// ---------------------------------------------------------------------------
// Types for CDNA5 WMMA
// ---------------------------------------------------------------------------
typedef __attribute__((ext_vector_type(16))) __bf16 v16bf;
typedef __attribute__((ext_vector_type(8)))  __bf16 v8bf;
typedef __attribute__((ext_vector_type(8)))  float  v8f;

#define NODES 30000
#define EDGES 180000
#define MTILE 128           // block M tile (8 waves as 4x2 of 32x32 wave tiles)
#define NTILE 64
#define MP    30080         // roundup(30000, 128)

static inline int rup(int x, int m) { return (x + m - 1) / m * m; }

// ---------------------------------------------------------------------------
// Helpers
// ---------------------------------------------------------------------------
__device__ inline __bf16 to_bf16(float f) {
  unsigned u = __float_as_uint(f);
  unsigned r = u + 0x7FFFu + ((u >> 16) & 1u);   // round-to-nearest-even
  unsigned short h = (unsigned short)(r >> 16);
  return __builtin_bit_cast(__bf16, h);
}

__device__ inline void atomicMaxF32(float* addr, float val) {
  unsigned* ua = (unsigned*)addr;
  unsigned old = *ua;
  while (true) {
    float f = __uint_as_float(old);
    if (f >= val) return;
    unsigned assumed = old;
    old = atomicCAS(ua, assumed, __float_as_uint(val));
    if (old == assumed) return;
  }
}

// ---------------------------------------------------------------------------
// Conversion kernels: f32 -> bf16 with zero padding
// ---------------------------------------------------------------------------
__global__ void cvt_a_kernel(const float* __restrict__ X, __bf16* __restrict__ A,
                             int M, int K, int Kp, int Mp) {
  size_t total = (size_t)Mp * (size_t)Kp;
  for (size_t idx = (size_t)blockIdx.x * blockDim.x + threadIdx.x; idx < total;
       idx += (size_t)gridDim.x * blockDim.x) {
    int m = (int)(idx / (size_t)Kp);
    int k = (int)(idx % (size_t)Kp);
    float f = (m < M && k < K) ? X[(size_t)m * K + k] : 0.0f;
    A[idx] = to_bf16(f);
  }
}

// W: [K x N] row-major -> Wt: [Np x Kp] row-major (transposed, zero padded)
__global__ void cvt_w_kernel(const float* __restrict__ W, __bf16* __restrict__ Wt,
                             int K, int N, int Kp, int Np) {
  size_t total = (size_t)Np * (size_t)Kp;
  for (size_t idx = (size_t)blockIdx.x * blockDim.x + threadIdx.x; idx < total;
       idx += (size_t)gridDim.x * blockDim.x) {
    int n = (int)(idx / (size_t)Kp);
    int k = (int)(idx % (size_t)Kp);
    float f = (n < N && k < K) ? W[(size_t)k * N + n] : 0.0f;
    Wt[idx] = to_bf16(f);
  }
}

// ---------------------------------------------------------------------------
// Fragment loads (16-bit WMMA operand layouts, wave32)
//   A 16x32: lanes 0-15  -> e0..7 = K k..k+7,   e8..15 = K k+16..k+23
//            lanes 16-31 -> e0..7 = K k+8..k+15, e8..15 = K k+24..k+31
//   B 32x16: same K order as A (lanes 0-15: e0..7 = K k..k+7, e8..15 = K k+16..k+23;
//            lanes 16-31: e0..7 = K k+8..k+15, e8..15 = K k+24..k+31)
// ---------------------------------------------------------------------------
__device__ inline void load_frag_a(const __bf16* __restrict__ p, int k, int half,
                                   v16bf& f) {
  v8bf lo = *(const v8bf*)(p + k + half * 8);
  v8bf hi = *(const v8bf*)(p + k + 16 + half * 8);
#pragma unroll
  for (int i = 0; i < 8; ++i) { f[i] = lo[i]; f[8 + i] = hi[i]; }
}

__device__ inline void load_frag_b(const __bf16* __restrict__ p, int k, int half,
                                   v16bf& f) {
  v8bf lo = *(const v8bf*)(p + k + half * 8);
  v8bf hi = *(const v8bf*)(p + k + 16 + half * 8);
#pragma unroll
  for (int i = 0; i < 8; ++i) { f[i] = lo[i]; f[8 + i] = hi[i]; }
}

// ---------------------------------------------------------------------------
// bf16 WMMA GEMM:  C[M x N] = A[Mp x Kp] * Bt[Np x Kp]^T + bias
// 256 threads = 8 waves arranged 4(M) x 2(N); wave tile 32x32 (2x2 WMMA).
// Register double-buffered K pipeline (Kp must be a multiple of 64).
// ---------------------------------------------------------------------------
__global__ __launch_bounds__(256)
void gemm_bf16_wmma(const __bf16* __restrict__ A, const __bf16* __restrict__ Bt,
                    const float* __restrict__ bias, float* __restrict__ C,
                    int M, int N, int Kp) {
  const int lane = threadIdx.x & 31;
  const int wave = threadIdx.x >> 5;
  const int half = lane >> 4;
  const int l16  = lane & 15;
  const int mi   = wave & 3;          // 0..3
  const int ni   = wave >> 2;         // 0..1

  const int row0 = blockIdx.x * MTILE + mi * 32;
  const int col0 = blockIdx.y * NTILE + ni * 32;

  const __bf16* a0 = A  + (size_t)(row0 + l16) * Kp;
  const __bf16* a1 = A  + (size_t)(row0 + 16 + l16) * Kp;
  const __bf16* b0 = Bt + (size_t)(col0 + l16) * Kp;
  const __bf16* b1 = Bt + (size_t)(col0 + 16 + l16) * Kp;

  v8f acc00 = (v8f){0.f,0.f,0.f,0.f,0.f,0.f,0.f,0.f};
  v8f acc01 = acc00, acc10 = acc00, acc11 = acc00;

  // Ping-pong fragment sets
  v16bf fa0[2], fa1[2], fb0[2], fb1[2];

  load_frag_a(a0, 0, half, fa0[0]);
  load_frag_a(a1, 0, half, fa1[0]);
  load_frag_b(b0, 0, half, fb0[0]);
  load_frag_b(b1, 0, half, fb1[0]);
  load_frag_a(a0, 32, half, fa0[1]);
  load_frag_a(a1, 32, half, fa1[1]);
  load_frag_b(b0, 32, half, fb0[1]);
  load_frag_b(b1, 32, half, fb1[1]);

#define WMMA4(p)                                                               \
  do {                                                                         \
    acc00 = __builtin_amdgcn_wmma_f32_16x16x32_bf16(false, fa0[p], false,      \
                fb0[p], (short)0, acc00, false, false);                        \
    acc01 = __builtin_amdgcn_wmma_f32_16x16x32_bf16(false, fa0[p], false,      \
                fb1[p], (short)0, acc01, false, false);                        \
    acc10 = __builtin_amdgcn_wmma_f32_16x16x32_bf16(false, fa1[p], false,      \
                fb0[p], (short)0, acc10, false, false);                        \
    acc11 = __builtin_amdgcn_wmma_f32_16x16x32_bf16(false, fa1[p], false,      \
                fb1[p], (short)0, acc11, false, false);                        \
  } while (0)

  // Steady state: consume set p while the other set's loads (issued one full
  // iteration earlier) are in flight.
  int k = 0;
  for (; k + 64 < Kp; k += 64) {
    WMMA4(0);
    load_frag_a(a0, k + 64, half, fa0[0]);
    load_frag_a(a1, k + 64, half, fa1[0]);
    load_frag_b(b0, k + 64, half, fb0[0]);
    load_frag_b(b1, k + 64, half, fb1[0]);
    WMMA4(1);
    load_frag_a(a0, k + 96, half, fa0[1]);
    load_frag_a(a1, k + 96, half, fa1[1]);
    load_frag_b(b0, k + 96, half, fb0[1]);
    load_frag_b(b1, k + 96, half, fb1[1]);
  }
  WMMA4(0);
  WMMA4(1);
#undef WMMA4

  // Store: acc VGPR i holds row (i + half*8) within the 16-row tile,
  // column = tile col base + l16.
  const v8f* accs[4] = {&acc00, &acc01, &acc10, &acc11};
#pragma unroll
  for (int t = 0; t < 4; ++t) {
    int col = col0 + (t & 1) * 16 + l16;
    int rbase = row0 + (t >> 1) * 16 + half * 8;
    if (col >= N) continue;
    float bv = bias[col];
    v8f acc = *accs[t];
#pragma unroll
    for (int i = 0; i < 8; ++i) {
      int r = rbase + i;
      if (r < M) C[(size_t)r * N + col] = acc[i] + bv;
    }
  }
}

// ---------------------------------------------------------------------------
// Edge / attention kernels
// ---------------------------------------------------------------------------
__global__ void fill_f32(float* __restrict__ p, size_t n, float v) {
  for (size_t i = (size_t)blockIdx.x * blockDim.x + threadIdx.x; i < n;
       i += (size_t)gridDim.x * blockDim.x)
    p[i] = v;
}

// scores[e] = dot(q[dst], k[src]) / sqrt(d)   (one wave32 per edge)
__global__ void edge_scores_kernel(const float* __restrict__ q,
                                   const float* __restrict__ k,
                                   const int* __restrict__ ei, int E, int d,
                                   float* __restrict__ scores) {
  int e = blockIdx.x * (blockDim.x >> 5) + (threadIdx.x >> 5);
  if (e >= E) return;
  int lane = threadIdx.x & 31;
  int s = ei[e];
  int t = ei[E + e];
  const float* qd = q + (size_t)t * d;
  const float* kd = k + (size_t)s * d;
  float acc = 0.f;
  for (int i = lane; i < d; i += 32) acc += qd[i] * kd[i];
#pragma unroll
  for (int off = 16; off > 0; off >>= 1) acc += __shfl_down(acc, off, 32);
  if (lane == 0) scores[e] = acc * rsqrtf((float)d);
}

__global__ void edge_max_kernel(const float* __restrict__ scores,
                                const int* __restrict__ ei, int E,
                                float* __restrict__ m) {
  int e = blockIdx.x * blockDim.x + threadIdx.x;
  if (e >= E) return;
  atomicMaxF32(&m[ei[E + e]], scores[e]);
}

__global__ void edge_expsum_kernel(float* __restrict__ scores,
                                   const int* __restrict__ ei, int E,
                                   const float* __restrict__ m,
                                   float* __restrict__ ssum) {
  int e = blockIdx.x * blockDim.x + threadIdx.x;
  if (e >= E) return;
  int t = ei[E + e];
  float v = __expf(scores[e] - m[t]);
  scores[e] = v;
  atomicAdd(&ssum[t], v);
}

// dest[dst] += alpha * src_mat[src]   (one wave32 per edge)
__global__ void edge_aggregate_kernel(const float* __restrict__ escores,
                                      const float* __restrict__ ssum,
                                      const float* __restrict__ src_mat,
                                      const int* __restrict__ ei, int E, int d,
                                      float* __restrict__ dest) {
  int e = blockIdx.x * (blockDim.x >> 5) + (threadIdx.x >> 5);
  if (e >= E) return;
  int lane = threadIdx.x & 31;
  int s = ei[e];
  int t = ei[E + e];
  float alpha = escores[e] / (ssum[t] + 1e-16f);
  const float* vs = src_mat + (size_t)s * d;
  float* dd = dest + (size_t)t * d;
  for (int i = lane; i < d; i += 32) atomicAdd(&dd[i], alpha * vs[i]);
}

__global__ void elu_kernel(float* __restrict__ p, size_t n) {
  for (size_t i = (size_t)blockIdx.x * blockDim.x + threadIdx.x; i < n;
       i += (size_t)gridDim.x * blockDim.x) {
    float x = p[i];
    p[i] = x > 0.f ? x : expm1f(x);
  }
}

// dest = dest(skip) + attn + bv * (node has in-edges); optional ELU
__global__ void epilogue_agg_kernel(float* __restrict__ dest,
                                    const float* __restrict__ attn,
                                    const float* __restrict__ bv,
                                    const float* __restrict__ ssum,
                                    int M, int N, int do_elu) {
  size_t total = (size_t)M * N;
  for (size_t i = (size_t)blockIdx.x * blockDim.x + threadIdx.x; i < total;
       i += (size_t)gridDim.x * blockDim.x) {
    int r = (int)(i / (size_t)N);
    int c = (int)(i % (size_t)N);
    float x = dest[i] + attn[i] + (ssum[r] > 0.f ? bv[c] : 0.f);
    dest[i] = do_elu ? (x > 0.f ? x : expm1f(x)) : x;
  }
}

// concat[m, 0:64] = a[m]; concat[m, 64:128] = b[m]
__global__ void concat_kernel(const float* __restrict__ a,
                              const float* __restrict__ b,
                              float* __restrict__ c, int M) {
  size_t total = (size_t)M * 128;
  for (size_t i = (size_t)blockIdx.x * blockDim.x + threadIdx.x; i < total;
       i += (size_t)gridDim.x * blockDim.x) {
    int m = (int)(i >> 7);
    int ccol = (int)(i & 127);
    c[i] = (ccol < 64) ? a[(size_t)m * 64 + ccol] : b[(size_t)m * 64 + (ccol - 64)];
  }
}

// ---------------------------------------------------------------------------
// Host orchestration
// ---------------------------------------------------------------------------
namespace {

struct Scratch {
  __bf16* abuf;    // MP x 3008 bf16
  __bf16* wbuf;    // 3008 x 3008 bf16 (Np x Kp)
  float*  qbuf;    // 30000 x 3000 f32
  float*  kbuf;    // 30000 x 3000 f32 (also reused as f32 aggX)
  float*  scores;  // EDGES
  float*  mmax;    // NODES
  float*  ssum;    // NODES
  float*  zbias;   // 3008 zeros
};

struct LayerW {
  const float *Wq, *bq, *Wk, *bk, *Wv, *bv, *Ws, *bs;
};

static void run_gemm(const __bf16* A, const __bf16* Bt, const float* bias,
                     float* C, int M, int N, int Kp, hipStream_t stream) {
  dim3 grid((M + MTILE - 1) / MTILE, rup(N, NTILE) / NTILE);
  gemm_bf16_wmma<<<grid, 256, 0, stream>>>(A, Bt, bias, C, M, N, Kp);
}

static void run_layer(const float* X, int din, int dout, const LayerW& p,
                      const int* ei, Scratch& s, float* dest, bool do_elu,
                      hipStream_t stream) {
  const int Kp = rup(din, 64);     // pipeline requires multiple of 64
  const int Np = rup(dout, NTILE);
  const int GS = 2048;
  const bool agg_first = (din < dout);   // scatter in the smaller dimension

  // X -> bf16 (padded)
  cvt_a_kernel<<<GS, 256, 0, stream>>>(X, s.abuf, NODES, din, Kp, MP);

  // Q, K projections (needed for attention scores in d_out dims)
  cvt_w_kernel<<<GS, 256, 0, stream>>>(p.Wq, s.wbuf, din, dout, Kp, Np);
  run_gemm(s.abuf, s.wbuf, p.bq, s.qbuf, NODES, dout, Kp, stream);
  cvt_w_kernel<<<GS, 256, 0, stream>>>(p.Wk, s.wbuf, din, dout, Kp, Np);
  run_gemm(s.abuf, s.wbuf, p.bk, s.kbuf, NODES, dout, Kp, stream);

  // Segment softmax over edges
  edge_scores_kernel<<<(EDGES + 7) / 8, 256, 0, stream>>>(
      s.qbuf, s.kbuf, ei, EDGES, dout, s.scores);
  fill_f32<<<(NODES + 255) / 256, 256, 0, stream>>>(s.mmax, NODES,
                                                    -__builtin_inff());
  fill_f32<<<(NODES + 255) / 256, 256, 0, stream>>>(s.ssum, NODES, 0.f);
  edge_max_kernel<<<(EDGES + 255) / 256, 256, 0, stream>>>(s.scores, ei, EDGES,
                                                           s.mmax);
  edge_expsum_kernel<<<(EDGES + 255) / 256, 256, 0, stream>>>(
      s.scores, ei, EDGES, s.mmax, s.ssum);

  // Skip/root: dest = X*Ws + bs  (initializes every dest element)
  cvt_w_kernel<<<GS, 256, 0, stream>>>(p.Ws, s.wbuf, din, dout, Kp, Np);
  run_gemm(s.abuf, s.wbuf, p.bs, dest, NODES, dout, Kp, stream);

  if (!agg_first) {
    // V = X*Wv + bv (overwrites qbuf), then scatter alpha*V (d_out dims)
    cvt_w_kernel<<<GS, 256, 0, stream>>>(p.Wv, s.wbuf, din, dout, Kp, Np);
    run_gemm(s.abuf, s.wbuf, p.bv, s.qbuf, NODES, dout, Kp, stream);
    edge_aggregate_kernel<<<(EDGES + 7) / 8, 256, 0, stream>>>(
        s.scores, s.ssum, s.qbuf, ei, EDGES, dout, dest);
    if (do_elu)
      elu_kernel<<<2048, 256, 0, stream>>>(dest, (size_t)NODES * dout);
  } else {
    // aggX[dst] = sum alpha * X[src]  (scatter in d_in dims, d_in < d_out)
    fill_f32<<<GS, 256, 0, stream>>>(s.kbuf, (size_t)NODES * din, 0.f);
    edge_aggregate_kernel<<<(EDGES + 7) / 8, 256, 0, stream>>>(
        s.scores, s.ssum, X, ei, EDGES, din, s.kbuf);
    // attn = aggX * Wv   (bias handled in epilogue: + bv iff node has edges)
    cvt_a_kernel<<<GS, 256, 0, stream>>>(s.kbuf, s.abuf, NODES, din, Kp, MP);
    cvt_w_kernel<<<GS, 256, 0, stream>>>(p.Wv, s.wbuf, din, dout, Kp, Np);
    run_gemm(s.abuf, s.wbuf, s.zbias, s.qbuf, NODES, dout, Kp, stream);
    epilogue_agg_kernel<<<GS, 256, 0, stream>>>(dest, s.qbuf, p.bv, s.ssum,
                                                NODES, dout, do_elu ? 1 : 0);
  }
}

}  // namespace

extern "C" void kernel_launch(void* const* d_in, const int* in_sizes, int n_in,
                              void* d_out, int out_size, void* d_ws, size_t ws_size,
                              hipStream_t stream) {
  const float* features = (const float*)d_in[0];   // [30000 x 3000]
  const float* img_feat = (const float*)d_in[1];   // [30000 x 1024]
  const int*   ei       = (const int*)d_in[2];     // [2 x 180000]

  LayerW L[12];
  for (int i = 0; i < 12; ++i) {
    const int b = 3 + i * 8;
    L[i].Wq = (const float*)d_in[b + 0]; L[i].bq = (const float*)d_in[b + 1];
    L[i].Wk = (const float*)d_in[b + 2]; L[i].bk = (const float*)d_in[b + 3];
    L[i].Wv = (const float*)d_in[b + 4]; L[i].bv = (const float*)d_in[b + 5];
    L[i].Ws = (const float*)d_in[b + 6]; L[i].bs = (const float*)d_in[b + 7];
  }
  enum { CONV1, CONV2, CONV3, CONV4, ICONV1, ICONV2, ICONV3, ICONV4,
         NECK, NECK2, C3, C4 };

  // Workspace carve-up (256B aligned)
  char* base = (char*)d_ws;
  size_t off = 0;
  auto carve = [&](size_t bytes) {
    void* p = base + off;
    off = (off + bytes + 255) & ~(size_t)255;
    return p;
  };
  Scratch s;
  s.abuf   = (__bf16*)carve((size_t)MP * 3008 * sizeof(__bf16));
  s.wbuf   = (__bf16*)carve((size_t)3008 * 3008 * sizeof(__bf16));
  s.qbuf   = (float*)carve((size_t)NODES * 3000 * sizeof(float));
  s.kbuf   = (float*)carve((size_t)NODES * 3000 * sizeof(float));
  s.scores = (float*)carve((size_t)EDGES * sizeof(float));
  s.mmax   = (float*)carve((size_t)NODES * sizeof(float));
  s.ssum   = (float*)carve((size_t)NODES * sizeof(float));
  s.zbias  = (float*)carve((size_t)3008 * sizeof(float));
  float* hbuf    = (float*)carve((size_t)NODES * 512 * sizeof(float));
  float* concatb = (float*)carve((size_t)NODES * 128 * sizeof(float));
  float* combine = (float*)carve((size_t)NODES * 64 * sizeof(float));
  (void)ws_size; (void)in_sizes; (void)n_in; (void)out_size;

  fill_f32<<<16, 256, 0, stream>>>(s.zbias, 3008, 0.f);

  // Output regions: (h2, img2, c2, h4, img4, c4)
  float* out    = (float*)d_out;
  float* o_h2   = out;
  float* o_img2 = out + (size_t)NODES * 64;
  float* o_c2   = out + (size_t)NODES * 64 * 2;
  float* o_h4   = out + (size_t)NODES * 64 * 3;
  float* o_img4 = o_h4 + (size_t)NODES * 3000;
  float* o_c4   = o_img4 + (size_t)NODES * 3000;

  // h chain
  run_layer(features, 3000, 512, L[CONV1], ei, s, hbuf,  true,  stream);  // h1
  run_layer(hbuf,      512,  64, L[CONV2], ei, s, o_h2,  false, stream);  // h2
  run_layer(o_h2,       64, 512, L[CONV3], ei, s, hbuf,  true,  stream);  // h3
  run_layer(hbuf,      512,3000, L[CONV4], ei, s, o_h4,  false, stream);  // h4

  // img chain
  run_layer(img_feat, 1024, 512, L[ICONV1], ei, s, hbuf,   true,  stream); // img1
  run_layer(hbuf,      512,  64, L[ICONV2], ei, s, o_img2, false, stream); // img2
  run_layer(o_img2,     64, 512, L[ICONV3], ei, s, hbuf,   true,  stream); // img3
  run_layer(hbuf,      512,3000, L[ICONV4], ei, s, o_img4, false, stream); // img4

  // neck chain
  concat_kernel<<<2048, 256, 0, stream>>>(o_h2, o_img2, concatb, NODES);
  run_layer(concatb, 128,  64, L[NECK],  ei, s, combine, true,  stream);   // combine
  run_layer(combine,  64,  64, L[NECK2], ei, s, o_c2,    false, stream);   // c2
  run_layer(o_c2,     64, 512, L[C3],    ei, s, hbuf,    true,  stream);   // c3a
  run_layer(hbuf,    512,3000, L[C4],    ei, s, o_c4,    false, stream);   // c4
}
